// MultiHeadAttention_31044023615778
// MI455X (gfx1250) — matrix operand order repaired and checked by reading, not compiled, not run on hardware
//
#include <hip/hip_runtime.h>
#include <hip/hip_bf16.h>

// ---------------------------------------------------------------------------
// MI455X (gfx1250) multi-head attention, bf16 WMMA (f32 accumulate).
// Compute-bound (~137 GFLOP vs ~130MB HBM traffic) -> all GEMMs through
// v_wmma_f32_16x16x32_bf16. Wave32; ds_swizzle reductions; TDM (tensor_load_
// to_lds) stages K/mask tiles in the attention kernel.
// ---------------------------------------------------------------------------

typedef __attribute__((ext_vector_type(16))) __bf16 bf16x16;
typedef __attribute__((ext_vector_type(8)))  __bf16 bf16x8;
typedef __attribute__((ext_vector_type(8)))  float  f32x8;
typedef __attribute__((ext_vector_type(4)))  unsigned int u32x4;
typedef __attribute__((ext_vector_type(4)))  int    i32x4;
typedef __attribute__((ext_vector_type(8)))  int    i32x8;

#define BATCH 4
#define SEQ   2048
#define EMB   1024
#define NH    16
#define HD    64
#define MDIM  (BATCH * SEQ)      /* 8192 */
#define ATTN_SCALE 0.125f        /* 64^-0.5 */

__device__ __forceinline__ unsigned short f2bf(float f) {
    unsigned int u = __float_as_uint(f);
    unsigned int r = (u + 0x7FFFu + ((u >> 16) & 1u)) >> 16;
    return (unsigned short)r;
}

__device__ __forceinline__ bf16x16 frag_cat(bf16x8 lo, bf16x8 hi) {
    return __builtin_shufflevector(lo, hi, 0, 1, 2, 3, 4, 5, 6, 7,
                                           8, 9, 10, 11, 12, 13, 14, 15);
}

// 16-lane (half-wave) reductions via ds_swizzle (wave32-safe; xor<=8 stays
// inside each 16-lane group that shares one WMMA C-row set).
__device__ __forceinline__ float red16_max(float v) {
    v = fmaxf(v, __uint_as_float((unsigned)__builtin_amdgcn_ds_swizzle(__float_as_uint(v), 0x041F)));
    v = fmaxf(v, __uint_as_float((unsigned)__builtin_amdgcn_ds_swizzle(__float_as_uint(v), 0x081F)));
    v = fmaxf(v, __uint_as_float((unsigned)__builtin_amdgcn_ds_swizzle(__float_as_uint(v), 0x101F)));
    v = fmaxf(v, __uint_as_float((unsigned)__builtin_amdgcn_ds_swizzle(__float_as_uint(v), 0x201F)));
    return v;
}
__device__ __forceinline__ float red16_sum(float v) {
    v += __uint_as_float((unsigned)__builtin_amdgcn_ds_swizzle(__float_as_uint(v), 0x041F));
    v += __uint_as_float((unsigned)__builtin_amdgcn_ds_swizzle(__float_as_uint(v), 0x081F));
    v += __uint_as_float((unsigned)__builtin_amdgcn_ds_swizzle(__float_as_uint(v), 0x101F));
    v += __uint_as_float((unsigned)__builtin_amdgcn_ds_swizzle(__float_as_uint(v), 0x201F));
    return v;
}

// ---------------------------------------------------------------------------
// TDM 2D tile load: global -> LDS with row padding (D# per ISA 08 §8.3/8.4).
// data_size_log: 0=1B,1=2B,2=4B.  pad_interval/pad_amount are field codes.
// Issue from one wave only; complete with s_wait_tensorcnt.
// 6-arg builtin form (clang-23 / therock-10.0 headers).
// ---------------------------------------------------------------------------
__device__ __forceinline__ void tdm_load_2d(unsigned lds_off, const void* gptr,
                                            unsigned data_size_log,
                                            unsigned tensor_d0, unsigned tensor_d1,
                                            unsigned long long d0_stride,
                                            unsigned tile_d0, unsigned tile_d1,
                                            unsigned pad_interval, unsigned pad_amount) {
    const unsigned long long ga = (unsigned long long)gptr;
    u32x4 g0;
    g0[0] = 1u;                                            // count=1, user mode
    g0[1] = lds_off;                                       // lds_addr
    g0[2] = (unsigned)(ga & 0xFFFFFFFFu);                  // global_addr lo
    g0[3] = (unsigned)((ga >> 32) & 0x01FFFFFFu) | (2u << 30);  // hi | type=2
    i32x8 g1;
    g1[0] = (int)((data_size_log << 16) | (1u << 20) |     // pad_enable
                  (pad_interval << 22) | (pad_amount << 25));
    g1[1] = (int)((tensor_d0 & 0xFFFFu) << 16);            // barrier_addr=0, dim0 lo
    g1[2] = (int)(((tensor_d0 >> 16) & 0xFFFFu) | ((tensor_d1 & 0xFFFFu) << 16));
    g1[3] = (int)(((tensor_d1 >> 16) & 0xFFFFu) | ((tile_d0 & 0xFFFFu) << 16));
    g1[4] = (int)(tile_d1 & 0xFFFFu);                      // tile_dim2 = 0
    g1[5] = (int)(d0_stride & 0xFFFFFFFFull);
    g1[6] = (int)((d0_stride >> 32) & 0xFFFFull);          // dim1_stride = 0 (2D)
    g1[7] = 0;
    const i32x4 gz4 = {0, 0, 0, 0};
    const i32x8 gz8 = {0, 0, 0, 0, 0, 0, 0, 0};
    __builtin_amdgcn_tensor_load_to_lds(g0, g1, gz4, gz4, gz8, 0);
}

// ---------------------------------------------------------------------------
// GEMM: Out[m][n] = sum_k A[m][k] * W[n][k] + bias[n]
//   M=8192, N=1024, K=1024.  Block tile 128x64, 8 waves (4x2 of 32x32 wave
//   tiles), BK=64 -> 8 WMMAs per LDS stage / barrier pair.
// A_BF16: A is bf16 (attention output) vs f32 (converted while staging).
// OUT_MODE 0: plain f32 [M][N].  OUT_MODE 1: head-split bf16 [B][H][S][64].
// ---------------------------------------------------------------------------
template <bool A_BF16, int OUT_MODE>
__global__ __launch_bounds__(256) void gemm_kernel(
        const void* __restrict__ Aptr, const float* __restrict__ W,
        const float* __restrict__ bias, void* __restrict__ Out) {
    constexpr int K = EMB, N = EMB;
    constexpr int LDA = 72;                       // rows 144B apart (16B aligned)
    __shared__ alignas(16) unsigned short lAs[128 * LDA];
    __shared__ alignas(16) unsigned short lBs[64 * LDA];

    const int tid  = threadIdx.x;
    const int wave = tid >> 5;
    const int lane = tid & 31;
    const int l15  = lane & 15;
    const int hh   = lane >> 4;                   // lane half (K split)
    const int wm   = (wave & 3) * 32;
    const int wn   = (wave >> 2) * 32;
    const int m0   = blockIdx.x * 128;
    const int n0   = blockIdx.y * 64;

    f32x8 c[2][2] = {};

    for (int k0 = 0; k0 < K; k0 += 64) {
        if (k0 + 64 < K) {  // gentle L2 prefetch of next weight tile
            __builtin_prefetch(W + (size_t)(n0 + (tid & 63)) * K + k0 + 64);
        }
        // ---- stage A tile (128 x 64) as bf16 ----
        if (A_BF16) {
            const unsigned short* A = (const unsigned short*)Aptr;
            const int col = (tid & 7) * 8, rb = tid >> 3;
#pragma unroll
            for (int p = 0; p < 4; ++p) {
                const int row = rb + p * 32;
                *(uint4*)(&lAs[row * LDA + col]) =
                    *(const uint4*)(A + (size_t)(m0 + row) * K + k0 + col);
            }
        } else {
            const float* A = (const float*)Aptr;
            const int col = (tid & 15) * 4, rb = tid >> 4;
#pragma unroll
            for (int p = 0; p < 8; ++p) {
                const int row = rb + p * 16;
                const float4 v = *(const float4*)(A + (size_t)(m0 + row) * K + k0 + col);
                const unsigned p0 = (unsigned)f2bf(v.x) | ((unsigned)f2bf(v.y) << 16);
                const unsigned p1 = (unsigned)f2bf(v.z) | ((unsigned)f2bf(v.w) << 16);
                *(uint2*)(&lAs[row * LDA + col]) = make_uint2(p0, p1);
            }
        }
        // ---- stage W tile (64 x 64, rows = output features) as bf16 ----
        {
            const int col = (tid & 15) * 4, rb = tid >> 4;
#pragma unroll
            for (int p = 0; p < 4; ++p) {
                const int row = rb + p * 16;
                const float4 v = *(const float4*)(W + (size_t)(n0 + row) * K + k0 + col);
                const unsigned p0 = (unsigned)f2bf(v.x) | ((unsigned)f2bf(v.y) << 16);
                const unsigned p1 = (unsigned)f2bf(v.z) | ((unsigned)f2bf(v.w) << 16);
                *(uint2*)(&lBs[row * LDA + col]) = make_uint2(p0, p1);
            }
        }
        __syncthreads();

        // ---- fragments in contiguous k order (k = 16*hh + e) for A and B, 2 k-steps ----
#pragma unroll
        for (int ks = 0; ks < 2; ++ks) {
            bf16x16 af[2], bfm[2];
#pragma unroll
            for (int mi = 0; mi < 2; ++mi) {  // A: K = 16*hh + e (contiguous)
                const unsigned short* pr = &lAs[(wm + mi * 16 + l15) * LDA + ks * 32];
                af[mi] = frag_cat(*(const bf16x8*)(pr + hh * 16),
                                  *(const bf16x8*)(pr + hh * 16 + 8));
            }
#pragma unroll
            for (int ni = 0; ni < 2; ++ni) {  // B: K = 16*hh + e (contiguous)
                const unsigned short* pr = &lBs[(wn + ni * 16 + l15) * LDA + ks * 32 + hh * 16];
                bfm[ni] = frag_cat(*(const bf16x8*)(pr), *(const bf16x8*)(pr + 8));
            }
#pragma unroll
            for (int mi = 0; mi < 2; ++mi)
#pragma unroll
                for (int ni = 0; ni < 2; ++ni)
                    c[mi][ni] = __builtin_amdgcn_wmma_f32_16x16x32_bf16(
                        false, af[mi], false, bfm[ni], (short)0, c[mi][ni], false, false);
        }
        __syncthreads();
    }

    // ---- epilogue: C/D layout row = r + 8*hh, col = l15 ----
#pragma unroll
    for (int ni = 0; ni < 2; ++ni) {
        const int n  = n0 + wn + ni * 16 + l15;
        const float bv = bias[n];
#pragma unroll
        for (int mi = 0; mi < 2; ++mi) {
#pragma unroll
            for (int r = 0; r < 8; ++r) {
                const int m = m0 + wm + mi * 16 + r + hh * 8;
                const float val = c[mi][ni][r] + bv;
                if (OUT_MODE == 0) {
                    ((float*)Out)[(size_t)m * N + n] = val;
                } else {
                    const int h = n >> 6, d = n & 63;
                    const int bb = m >> 11, t = m & 2047;
                    ((unsigned short*)Out)[(((size_t)(bb * NH + h)) * SEQ + t) * HD + d] =
                        f2bf(val);
                }
            }
        }
    }
}

// ---------------------------------------------------------------------------
// Flash attention: one block = (b, h, 64 query rows); 4 waves x 16 rows.
// Streams S in 64-wide tiles.  K tile + mask tile staged by the Tensor Data
// Mover (padded rows), V transposed manually, P round-trips through LDS.
// ---------------------------------------------------------------------------
__global__ __launch_bounds__(128) void attn_kernel(
        const unsigned short* __restrict__ qh, const unsigned short* __restrict__ kh,
        const unsigned short* __restrict__ vh, const float* __restrict__ amask,
        const unsigned char* __restrict__ kpm, unsigned short* __restrict__ ao) {
    constexpr int LQ = 72;                        // bf16 tile stride (144B rows)
    constexpr int LM = 68;                        // f32 mask tile stride (272B rows)
    __shared__ alignas(16) unsigned short lQ[64 * LQ];
    __shared__ alignas(16) unsigned short lK[64 * LQ];
    __shared__ alignas(16) unsigned short lVt[64 * LQ];   // V transposed [d][s]
    __shared__ alignas(16) unsigned short lP[64 * LQ];
    __shared__ alignas(16) float lMask[64 * LM];

    const int tid  = threadIdx.x;
    const int wave = tid >> 5;
    const int lane = tid & 31;
    const int l15  = lane & 15;
    const int hh   = lane >> 4;

    const int t0 = blockIdx.x * 64;
    const int h  = blockIdx.y;
    const int b  = blockIdx.z;

    const unsigned short* Qg = qh + (((size_t)(b * NH + h)) * SEQ + t0) * HD;
    const unsigned short* Kg = kh + ((size_t)(b * NH + h)) * SEQ * HD;
    const unsigned short* Vg = vh + ((size_t)(b * NH + h)) * SEQ * HD;

    { // stage Q (64 x 64 bf16), once
        const int col = (tid & 7) * 8, rb = tid >> 3;
        for (int p = 0; p < 4; ++p) {
            const int row = rb + p * 16;
            *(uint4*)(&lQ[row * LQ + col]) = *(const uint4*)(Qg + (size_t)row * HD + col);
        }
    }
    __syncthreads();

    bf16x16 qf[2];
#pragma unroll
    for (int ks = 0; ks < 2; ++ks) {
        const unsigned short* pr = &lQ[(wave * 16 + l15) * LQ + ks * 32];
        qf[ks] = frag_cat(*(const bf16x8*)(pr + hh * 16),
                          *(const bf16x8*)(pr + hh * 16 + 8));
    }

    f32x8 o[4] = {};
    float mrow[8], lrow[8];
#pragma unroll
    for (int r = 0; r < 8; ++r) { mrow[r] = -__builtin_inff(); lrow[r] = 0.f; }

    for (int s0 = 0; s0 < SEQ; s0 += 64) {
        __syncthreads();                          // previous tile fully consumed
        if (wave == 0) {
            // K tile: 64x64 bf16, pad 4 DWORDs per 32-DWORD row -> stride 72.
            tdm_load_2d((unsigned)(unsigned long long)&lK[0],
                        Kg + (size_t)s0 * HD, 1u,
                        HD, SEQ, (unsigned long long)HD, HD, 64u,
                        4u /*32 dw*/, 3u /*4 dw*/);
            // mask tile: 64x64 f32, pad 4 DWORDs per 64-DWORD row -> stride 68.
            tdm_load_2d((unsigned)(unsigned long long)&lMask[0],
                        amask + (size_t)t0 * SEQ + s0, 2u,
                        SEQ, SEQ, (unsigned long long)SEQ, 64u, 64u,
                        5u /*64 dw*/, 3u /*4 dw*/);
        }
        { // stage V transposed (all 128 threads)
            const int col = (tid & 7) * 8, rb = tid >> 3;
            for (int p = 0; p < 4; ++p) {
                const int row = rb + p * 16;
                alignas(16) unsigned short tmp[8];
                *(uint4*)tmp = *(const uint4*)(Vg + (size_t)(s0 + row) * HD + col);
#pragma unroll
                for (int j = 0; j < 8; ++j) lVt[(col + j) * LQ + row] = tmp[j];
            }
        }
        if (wave == 0) __builtin_amdgcn_s_wait_tensorcnt(0);
        __syncthreads();

        // ---- scores = Q K^T (B frag: n = s row of lK, k = d contiguous) ----
        f32x8 sc[4];
#pragma unroll
        for (int ni = 0; ni < 4; ++ni) {
            f32x8 z = {};
#pragma unroll
            for (int ks = 0; ks < 2; ++ks) {
                const unsigned short* pr = &lK[(ni * 16 + l15) * LQ + ks * 32 + hh * 16];
                const bf16x16 kf = frag_cat(*(const bf16x8*)(pr), *(const bf16x8*)(pr + 8));
                z = __builtin_amdgcn_wmma_f32_16x16x32_bf16(
                        false, qf[ks], false, kf, (short)0, z, false, false);
            }
            sc[ni] = z;
        }

        // ---- scale + attn_mask + key_padding_mask ----
#pragma unroll
        for (int ni = 0; ni < 4; ++ni) {
            const int s = s0 + ni * 16 + l15;
            const float pad = kpm[(size_t)b * SEQ + s] ? -__builtin_inff() : 0.f;
#pragma unroll
            for (int r = 0; r < 8; ++r) {
                const float mv = lMask[(wave * 16 + r + hh * 8) * LM + ni * 16 + l15];
                sc[ni][r] = fmaf(sc[ni][r], ATTN_SCALE, mv) + pad;
            }
        }

        // ---- online softmax (row = r + 8*hh shared by each 16-lane group) ----
        float alpha[8];
#pragma unroll
        for (int r = 0; r < 8; ++r) {
            float tm = fmaxf(fmaxf(sc[0][r], sc[1][r]), fmaxf(sc[2][r], sc[3][r]));
            const float mnew = fmaxf(mrow[r], red16_max(tm));
            alpha[r] = (mrow[r] == -__builtin_inff()) ? 0.f : __expf(mrow[r] - mnew);
            mrow[r] = mnew;
            float rs = 0.f;
#pragma unroll
            for (int ni = 0; ni < 4; ++ni) {
                const float p = __expf(sc[ni][r] - mnew);
                sc[ni][r] = p;
                rs += p;
            }
            lrow[r] = lrow[r] * alpha[r] + red16_sum(rs);
        }

        // ---- P -> LDS (C layout -> A layout round trip), rescale O ----
#pragma unroll
        for (int ni = 0; ni < 4; ++ni)
#pragma unroll
            for (int r = 0; r < 8; ++r)
                lP[(wave * 16 + r + hh * 8) * LQ + ni * 16 + l15] = f2bf(sc[ni][r]);
#pragma unroll
        for (int oi = 0; oi < 4; ++oi)
#pragma unroll
            for (int r = 0; r < 8; ++r) o[oi][r] *= alpha[r];

        // ---- O += P V  (B frag from V^T rows: n = d, k = s contiguous) ----
#pragma unroll
        for (int ks = 0; ks < 2; ++ks) {
            const unsigned short* pr = &lP[(wave * 16 + l15) * LQ + ks * 32];
            const bf16x16 pf = frag_cat(*(const bf16x8*)(pr + hh * 16),
                                        *(const bf16x8*)(pr + hh * 16 + 8));
#pragma unroll
            for (int oi = 0; oi < 4; ++oi) {
                const unsigned short* vr = &lVt[(oi * 16 + l15) * LQ + ks * 32 + hh * 16];
                const bf16x16 vf = frag_cat(*(const bf16x8*)(vr), *(const bf16x8*)(vr + 8));
                o[oi] = __builtin_amdgcn_wmma_f32_16x16x32_bf16(
                            false, pf, false, vf, (short)0, o[oi], false, false);
            }
        }
    }

    // ---- normalize + write (B,T,C) bf16 for the output projection ----
#pragma unroll
    for (int oi = 0; oi < 4; ++oi) {
        const int d = oi * 16 + l15;
#pragma unroll
        for (int r = 0; r < 8; ++r) {
            const int t = t0 + wave * 16 + r + hh * 8;
            ao[(((size_t)b * SEQ + t) * EMB) + h * HD + d] = f2bf(o[oi][r] / lrow[r]);
        }
    }
}

// ---------------------------------------------------------------------------
extern "C" void kernel_launch(void* const* d_in, const int* in_sizes, int n_in,
                              void* d_out, int out_size, void* d_ws, size_t ws_size,
                              hipStream_t stream) {
    (void)in_sizes; (void)n_in; (void)out_size; (void)ws_size;
    const float*         query = (const float*)d_in[0];
    const float*         key_  = (const float*)d_in[1];
    const float*         value = (const float*)d_in[2];
    const float*         amask = (const float*)d_in[3];
    const unsigned char* kpm   = (const unsigned char*)d_in[4];
    const float* Wq = (const float*)d_in[5];  const float* bq = (const float*)d_in[6];
    const float* Wk = (const float*)d_in[7];  const float* bk = (const float*)d_in[8];
    const float* Wv = (const float*)d_in[9];  const float* bv = (const float*)d_in[10];
    const float* Wo = (const float*)d_in[11]; const float* bo = (const float*)d_in[12];

    const size_t headElems = (size_t)BATCH * NH * SEQ * HD;  // 8M bf16 each
    unsigned short* qh = (unsigned short*)d_ws;
    unsigned short* kh = qh + headElems;
    unsigned short* vh = kh + headElems;
    unsigned short* ao = vh + headElems;

    const dim3 gg(MDIM / 128, EMB / 64, 1);
    gemm_kernel<false, 1><<<gg, 256, 0, stream>>>(query, Wq, bq, qh);
    gemm_kernel<false, 1><<<gg, 256, 0, stream>>>(key_,  Wk, bk, kh);
    gemm_kernel<false, 1><<<gg, 256, 0, stream>>>(value, Wv, bv, vh);
    attn_kernel<<<dim3(SEQ / 64, NH, BATCH), 128, 0, stream>>>(qh, kh, vh, amask, kpm, ao);
    gemm_kernel<true, 0><<<gg, 256, 0, stream>>>(ao, Wo, bo, (float*)d_out);
}
